// LSTM_18098992185422
// MI455X (gfx1250) — matrix operand order repaired and checked by reading, not compiled, not run on hardware
//
#include <hip/hip_runtime.h>
#include <hip/hip_bf16.h>

// ---------------------------------------------------------------------------
// LSTM (B=128, T=2048, I=64, H=128) for gfx1250 / MI455X.
//
// 8 workgroups (one 16-row batch slice each, no inter-WG sync), 8 waves per
// workgroup; wave j owns hidden columns [16j,16j+16) and computes its own
// i/f/g/o gate tiles with v_wmma_f32_16x16x32_f16, so the LSTM cell update is
// wave-local in the WMMA C/D layout. Weights pinned in VGPRs (24 v16h
// B-fragments/wave). c stays in registers.
//
// h is DOUBLE-BUFFERED in LDS: step t reads h_buf[t&1], writes h_buf[~t&1],
// so one barrier round per timestep is enough (signal once own LDS ops have
// retired, wait only after the out-stores are issued). Activations use
// v_tanh_f32 (gfx1250 TRANS op).
// ---------------------------------------------------------------------------

typedef __attribute__((ext_vector_type(16))) _Float16 v16h;
typedef __attribute__((ext_vector_type(8)))  float    v8f;
typedef __attribute__((ext_vector_type(4)))  float    v4f;

#define B_  128
#define T_  2048
#define I_  64
#define H_  128

// LDS row strides (bytes), padded 16B so 16 rows hit distinct banks.
#define HB_STRIDE 272   // 128 f16 (256B) + 16B pad  -> 68 dwords/row
#define XB_STRIDE 144   //  64 f16 (128B) + 16B pad  -> 36 dwords/row

union AFrag { v16h v; uint4 q[2]; };

// ---- fast activations ------------------------------------------------------
#if __has_builtin(__builtin_amdgcn_rcpf)
#define RCPF(x) __builtin_amdgcn_rcpf(x)
#else
#define RCPF(x) (1.0f / (x))
#endif

#if __has_builtin(__builtin_amdgcn_tanhf)
__device__ __forceinline__ float tanh_fast(float x) {
    return __builtin_amdgcn_tanhf(x);              // v_tanh_f32
}
__device__ __forceinline__ float sigmoid_fast(float x) {
    return fmaf(0.5f, __builtin_amdgcn_tanhf(0.5f * x), 0.5f);
}
#else
__device__ __forceinline__ float tanh_fast(float x) {
    float e = __expf(-2.0f * fabsf(x));
    float r = (1.0f - e) * RCPF(1.0f + e);
    return copysignf(r, x);
}
__device__ __forceinline__ float sigmoid_fast(float x) {
    return RCPF(1.0f + __expf(-x));
}
#endif

// ---- split workgroup barrier ----------------------------------------------
__device__ __forceinline__ void wg_arrive_after_lds_ops() {
    // own LDS reads AND writes retired (in-order per wave) -> arrive
    asm volatile("s_wait_dscnt 0x0\n\ts_barrier_signal -1" ::: "memory");
}
__device__ __forceinline__ void wg_wait() {
    asm volatile("s_barrier_wait -1" ::: "memory");
}

__device__ __forceinline__ void cvt16(const float* __restrict__ p, v16h& w) {
    const v4f* pv = (const v4f*)p;                 // 64B aligned by layout
    #pragma unroll
    for (int k = 0; k < 4; ++k) {
        v4f q = pv[k];
        w[4 * k + 0] = (_Float16)q[0];
        w[4 * k + 1] = (_Float16)q[1];
        w[4 * k + 2] = (_Float16)q[2];
        w[4 * k + 3] = (_Float16)q[3];
    }
}

__global__ __launch_bounds__(256, 1)
void lstm_wmma_f16(const float* __restrict__ x,
                   const float* __restrict__ W_ih,
                   const float* __restrict__ W_hh,
                   const float* __restrict__ b_ih,
                   const float* __restrict__ b_hh,
                   float* __restrict__ out)
{
    __shared__ __align__(16) unsigned char h_buf[2][16 * HB_STRIDE];
    __shared__ __align__(16) unsigned char x_buf[2][16 * XB_STRIDE];

    const int tid   = threadIdx.x;
    const int lane  = tid & 31;
    const int wavej = tid >> 5;          // 0..7: hidden tile (cols 16j..)
    const int b0    = blockIdx.x * 16;   // batch slice base

    const int ncol  = lane & 15;         // N col (B/C/D) == M row (A)
    const int khalf = lane >> 4;         // 0: lanes 0-15, 1: lanes 16-31

    // ---- one-time: weight B-fragments + bias into registers ---------------
    // B[k,n] = W[n,k]; lane holds W[nbase+ncol, c*32 + khalf*16 .. +15],
    // 16 contiguous floats in row-major W.
    v16h whh[4][4];                      // [gate][k-chunk of H]
    v16h wih[4][2];                      // [gate][k-chunk of I]
    v8f  biasv[4];                       // read directly as WMMA C operand
    #pragma unroll
    for (int g = 0; g < 4; ++g) {
        const int n = g * H_ + wavej * 16 + ncol;
        const float bsum = b_ih[n] + b_hh[n];
        #pragma unroll
        for (int e = 0; e < 8; ++e) biasv[g][e] = bsum;
        #pragma unroll
        for (int c = 0; c < 4; ++c)
            cvt16(W_hh + (size_t)n * H_ + c * 32 + khalf * 16, whh[g][c]);
        #pragma unroll
        for (int c = 0; c < 2; ++c)
            cvt16(W_ih + (size_t)n * I_ + c * 32 + khalf * 16, wih[g][c]);
    }

    // ---- init: h0 = 0 (both buffers), stage x(t=0) ------------------------
    for (int i = tid; i < (2 * 16 * HB_STRIDE) / 4; i += 256)
        ((unsigned int*)h_buf)[i] = 0u;

    const int xr = tid >> 4;             // x stage: row 0..15
    const int xc = (tid & 15) * 4;       //          col {0,4,...,60}
    {
        const float* px = x + ((size_t)(b0 + xr) * T_ + 0) * I_ + xc;
        v4f v = *(const v4f*)px;
        _Float16* d = (_Float16*)(x_buf[0] + xr * XB_STRIDE + xc * 2);
        d[0] = (_Float16)v[0]; d[1] = (_Float16)v[1];
        d[2] = (_Float16)v[2]; d[3] = (_Float16)v[3];
    }

    v8f cstate;
    #pragma unroll
    for (int e = 0; e < 8; ++e) cstate[e] = 0.0f;

    __syncthreads();

    // ---- recurrence: ONE barrier round per timestep -----------------------
    for (int t = 0; t < T_; ++t) {
        const int cur = t & 1, nxt = cur ^ 1;

        // A-fragments from LDS. A order used here (not the ISA table order): lane (M=ncol) holds
        // 16 contiguous K: kbase=c*32+khalf*16 .. +15 (matches B).
        AFrag ah[4];
        #pragma unroll
        for (int c = 0; c < 4; ++c) {
            const unsigned char* base =
                h_buf[cur] + ncol * HB_STRIDE + (c * 32 + khalf * 16) * 2;
            ah[c].q[0] = *(const uint4*)(base);
            ah[c].q[1] = *(const uint4*)(base + 16);
        }
        AFrag ax[2];
        #pragma unroll
        for (int c = 0; c < 2; ++c) {
            const unsigned char* base =
                x_buf[cur] + ncol * XB_STRIDE + (c * 32 + khalf * 16) * 2;
            ax[c].q[0] = *(const uint4*)(base);
            ax[c].q[1] = *(const uint4*)(base + 16);
        }

        // Pull x(t+1) from HBM while WMMAs run; hint further ahead.
        const int t1 = (t + 1 < T_) ? t + 1 : t;          // clamp, no branch
        const int t8 = (t + 8 < T_) ? t + 8 : t;
        v4f xnv = *(const v4f*)(x + ((size_t)(b0 + xr) * T_ + t1) * I_ + xc);
        __builtin_prefetch(x + ((size_t)(b0 + xr) * T_ + t8) * I_ + xc, 0, 1);

        // 4 independent 6-WMMA accumulation chains (one per gate) for ILP;
        // bias enters as the C operand of the first WMMA (no moves).
        v8f acc[4];
        #pragma unroll
        for (int g = 0; g < 4; ++g) {
            v8f a = biasv[g];
            #pragma unroll
            for (int c = 0; c < 4; ++c)
                a = __builtin_amdgcn_wmma_f32_16x16x32_f16(
                        false, ah[c].v, false, whh[g][c], (short)0, a, false, false);
            #pragma unroll
            for (int c = 0; c < 2; ++c)
                a = __builtin_amdgcn_wmma_f32_16x16x32_f16(
                        false, ax[c].v, false, wih[g][c], (short)0, a, false, false);
            acc[g] = a;
        }

        // Pointwise cell update, wave-local in C/D layout:
        // VGPR e -> (m = e + 8*khalf, n = ncol) of this wave's hidden tile.
        v8f hnew;
        #pragma unroll
        for (int e = 0; e < 8; ++e) {
            float iv = sigmoid_fast(acc[0][e]);
            float fv = sigmoid_fast(acc[1][e]);
            float gv = tanh_fast(acc[2][e]);
            float ov = sigmoid_fast(acc[3][e]);
            float cn = fmaf(fv, cstate[e], iv * gv);
            cstate[e] = cn;
            hnew[e]   = ov * tanh_fast(cn);
        }

        // h_new -> the OTHER h buffer (no read/write aliasing this step);
        // stage x(t+1) into the other x buffer.
        const int hc = wavej * 16 + ncol;
        #pragma unroll
        for (int e = 0; e < 8; ++e) {
            const int m = e + khalf * 8;
            *(_Float16*)(h_buf[nxt] + m * HB_STRIDE + hc * 2) = (_Float16)hnew[e];
        }
        {
            _Float16* d = (_Float16*)(x_buf[nxt] + xr * XB_STRIDE + xc * 2);
            d[0] = (_Float16)xnv[0]; d[1] = (_Float16)xnv[1];
            d[2] = (_Float16)xnv[2]; d[3] = (_Float16)xnv[3];
        }

        // Single barrier round: arrive once own LDS ops retired, let the
        // out-stores issue during the barrier round-trip, then wait.
        wg_arrive_after_lds_ops();
        #pragma unroll
        for (int e = 0; e < 8; ++e) {
            const int m = e + khalf * 8;
            out[((size_t)(b0 + m) * T_ + t) * H_ + hc] = hnew[e];
        }
        wg_wait();
    }
}

extern "C" void kernel_launch(void* const* d_in, const int* in_sizes, int n_in,
                              void* d_out, int out_size, void* d_ws, size_t ws_size,
                              hipStream_t stream) {
    (void)in_sizes; (void)n_in; (void)out_size; (void)d_ws; (void)ws_size;
    const float* x    = (const float*)d_in[0];
    const float* W_ih = (const float*)d_in[1];
    const float* W_hh = (const float*)d_in[2];
    const float* b_ih = (const float*)d_in[3];
    const float* b_hh = (const float*)d_in[4];
    float* out = (float*)d_out;

    dim3 grid(B_ / 16);   // 8 workgroups: one 16-row batch slice each
    dim3 block(256);      // 8 waves: one hidden 16-col tile each
    hipLaunchKernelGGL(lstm_wmma_f16, grid, block, 0, stream,
                       x, W_ih, W_hh, b_ih, b_hh, out);
}
